// NET_41618233098683
// MI455X (gfx1250) — matrix operand order repaired and checked by reading, not compiled, not run on hardware
//
#include <hip/hip_runtime.h>
#include <hip/hip_bf16.h>

// ---------------------------------------------------------------------------
// out[64,4096] = x[64,25088] @ (centroids[labels]).T + bias
// HBM-bound by the 411 MB int32 label stream (~18 us @ 23.3 TB/s).
// v_wmma_f32_16x16x32_bf16 tiles; shfl-LUT dequant; K=64 double-buffered
// pipeline with a single barrier per step; x pre-converted to bf16 once.
// ---------------------------------------------------------------------------

typedef __attribute__((ext_vector_type(16))) __bf16 v16bf;
typedef __attribute__((ext_vector_type(8)))  float  v8f;
typedef __attribute__((ext_vector_type(4)))  int    v4i;
typedef __attribute__((ext_vector_type(4)))  float  v4f;
typedef __attribute__((ext_vector_type(2)))  unsigned v2u;

#define IN_DIM  25088
#define OUT_DIM 4096
#define B_DIM   64
#define KT2     64               // K per pipeline step (2 WMMA K-chunks)
#define NSTEP   (IN_DIM / KT2)   // 392
#define NT      32               // output-column tile per block
#define LDK     72               // padded LDS row (bf16) = 144 B, conflict-free

union Frag {
    v16bf v;
    v4i   q[2];
};

__device__ __forceinline__ unsigned short f32_to_bf16(float f) {
    unsigned u = __builtin_bit_cast(unsigned, f);
    unsigned rnd = 0x7FFFu + ((u >> 16) & 1u);   // round-to-nearest-even
    return (unsigned short)((u + rnd) >> 16);
}

__device__ __forceinline__ unsigned pk_bf16(float lo, float hi) {
#if __has_builtin(__builtin_amdgcn_cvt_pk_bf16_f32)
    typedef __attribute__((ext_vector_type(2))) __bf16 v2bf;
    v2bf r = __builtin_amdgcn_cvt_pk_bf16_f32(lo, hi);
    return __builtin_bit_cast(unsigned, r);
#else
    return (unsigned)f32_to_bf16(lo) | ((unsigned)f32_to_bf16(hi) << 16);
#endif
}

// one-shot x f32 -> bf16 (removes redundant per-block conversion work)
__global__ __launch_bounds__(256)
void convert_x_bf16(const float* __restrict__ x,
                    unsigned short* __restrict__ xbf, int n) {
    int i = (blockIdx.x * 256 + threadIdx.x) * 8;
    if (i >= n) return;
    v4f a = *(const v4f*)(x + i);
    v4f b = *(const v4f*)(x + i + 4);
    union { unsigned u[4]; v4i q; } r;
    r.u[0] = pk_bf16(a.x, a.y); r.u[1] = pk_bf16(a.z, a.w);
    r.u[2] = pk_bf16(b.x, b.y); r.u[3] = pk_bf16(b.z, b.w);
    *(v4i*)(xbf + i) = r.q;
}

template <bool PRECONV>
__global__ __launch_bounds__(256, 2)
void dequant_gemm_wmma(const float*          __restrict__ x,
                       const unsigned short* __restrict__ xbf,
                       const int*            __restrict__ labels,
                       const float*          __restrict__ centroids,
                       const float*          __restrict__ bias,
                       float*                __restrict__ out)
{
    // Double-buffered bf16 tiles: A = x (64x64), B = dequantized W (32x64)
    __shared__ __align__(16) unsigned short Abuf[2][B_DIM][LDK];
    __shared__ __align__(16) unsigned short Bbuf[2][NT][LDK];

    const int tid  = threadIdx.x;
    const int lane = tid & 31;
    const int wave = tid >> 5;
    const int n0   = blockIdx.x * NT;

    // Per-lane bf16 codebook: lane L holds centroids[L] (K == 32 == wave32)
    const int cbits = (int)f32_to_bf16(centroids[lane]);

    // Labels tile: 32 rows x 64 k ; 2x b128 per thread
    const int ln = tid >> 3;              // 0..31 row within n-tile
    const int lk = (tid & 7) * 4;         // 0..28 k chunk; second chunk at +32
    const int lab_base = (n0 + ln) * IN_DIM + lk;

    // x tile: 64 rows x 64 k ; 16 values per thread
    const int xm = tid >> 2;              // 0..63
    const int xk = (tid & 3) * 16;        // 0,16,32,48
    const int x_base = xm * IN_DIM + xk;

    // WMMA fragment mapping (ISA wave32 VGPR layouts)
    const int msub = (wave & 3) * 16;
    const int nsub = (wave >> 2) * 16;
    const int fr   = lane & 15;
    const int ka   = (lane >> 4) * 8;     // A: K base {0,8} (+16 block)
    const int kb   = (lane >> 4) * 16;    // unused: B loads use ka (same k order as A)

    v8f acc = {};

    v4i lab0, lab1;
    v4i xb0 = {}, xb1 = {};
    v4f xf0 = {}, xf1 = {}, xf2 = {}, xf3 = {};

    auto gload = [&](int k0) {
        lab0 = *(const v4i*)(labels + lab_base + k0);
        lab1 = *(const v4i*)(labels + lab_base + k0 + 32);
        if (PRECONV) {
            xb0 = *(const v4i*)(xbf + x_base + k0);
            xb1 = *(const v4i*)(xbf + x_base + k0 + 8);
        } else {
            xf0 = *(const v4f*)(x + x_base + k0);
            xf1 = *(const v4f*)(x + x_base + k0 + 4);
            xf2 = *(const v4f*)(x + x_base + k0 + 8);
            xf3 = *(const v4f*)(x + x_base + k0 + 12);
        }
    };

    auto dq = [&](int l) -> unsigned {   // label -> bf16 bits (hi 16 zero)
        return (unsigned)__builtin_amdgcn_ds_bpermute(l << 2, cbits);
    };

    auto stage = [&](int buf) {
        v2u p0, p1;
        p0.x = dq(lab0.x) | (dq(lab0.y) << 16);
        p0.y = dq(lab0.z) | (dq(lab0.w) << 16);
        p1.x = dq(lab1.x) | (dq(lab1.y) << 16);
        p1.y = dq(lab1.z) | (dq(lab1.w) << 16);
        *(v2u*)&Bbuf[buf][ln][lk]      = p0;   // ds_store_b64
        *(v2u*)&Bbuf[buf][ln][lk + 32] = p1;

        if (PRECONV) {
            *(v4i*)&Abuf[buf][xm][xk]     = xb0;  // pure copy, no VALU
            *(v4i*)&Abuf[buf][xm][xk + 8] = xb1;
        } else {
            union { unsigned u[4]; v4i q; } r0, r1;
            r0.u[0] = pk_bf16(xf0.x, xf0.y); r0.u[1] = pk_bf16(xf0.z, xf0.w);
            r0.u[2] = pk_bf16(xf1.x, xf1.y); r0.u[3] = pk_bf16(xf1.z, xf1.w);
            r1.u[0] = pk_bf16(xf2.x, xf2.y); r1.u[1] = pk_bf16(xf2.z, xf2.w);
            r1.u[2] = pk_bf16(xf3.x, xf3.y); r1.u[3] = pk_bf16(xf3.z, xf3.w);
            *(v4i*)&Abuf[buf][xm][xk]     = r0.q;
            *(v4i*)&Abuf[buf][xm][xk + 8] = r1.q;
        }
    };

    // ---- prologue: stage tile 0 ----
    gload(0);
    stage(0);
    __syncthreads();

    for (int ks = 0; ks < NSTEP; ++ks) {
        const int  cur  = ks & 1;
        const bool next = (ks + 1 < NSTEP);

        if (next) {
            gload((ks + 1) * KT2);
            if (ks + 4 < NSTEP)
                __builtin_prefetch(labels + lab_base + (ks + 4) * KT2, 0, 1);
        }

        // 2 WMMAs from current buffer (conflict-free b128 LDS reads)
#pragma unroll
        for (int j = 0; j < 2; ++j) {
            const int kj = j * 32;
            Frag a, b;
            a.q[0] = *(const v4i*)&Abuf[cur][msub + fr][kj + ka];
            a.q[1] = *(const v4i*)&Abuf[cur][msub + fr][kj + ka + 16];
            b.q[0] = *(const v4i*)&Bbuf[cur][nsub + fr][kj + ka];
            b.q[1] = *(const v4i*)&Bbuf[cur][nsub + fr][kj + ka + 16];
            acc = __builtin_amdgcn_wmma_f32_16x16x32_bf16(
                false, a.v, false, b.v, (short)0, acc, false, false);
        }

        if (next) stage(cur ^ 1);   // writes other buffer: no race with reads
        __syncthreads();            // single barrier per step
    }

    // ---- epilogue: C layout -> global, fused bias add ----
    const int orow = msub + (lane >> 4) * 8;   // VGPR r holds M = orow + r
    const int ocol = n0 + nsub + fr;
    const float bv = bias[ocol];
#pragma unroll
    for (int r = 0; r < 8; ++r) {
        out[(orow + r) * OUT_DIM + ocol] = acc[r] + bv;
    }
}

extern "C" void kernel_launch(void* const* d_in, const int* in_sizes, int n_in,
                              void* d_out, int out_size, void* d_ws, size_t ws_size,
                              hipStream_t stream) {
    const float* x         = (const float*)d_in[0];
    const int*   labels    = (const int*)d_in[1];
    const float* centroids = (const float*)d_in[2];
    const float* bias      = (const float*)d_in[3];
    float*       out       = (float*)d_out;

    const int    xn        = B_DIM * IN_DIM;
    const size_t xbf_bytes = (size_t)xn * sizeof(unsigned short);

    dim3 grid(OUT_DIM / NT);   // 128 blocks
    dim3 block(256);           // 8 wave32

    if (ws_size >= xbf_bytes) {
        unsigned short* xbf = (unsigned short*)d_ws;
        hipLaunchKernelGGL(convert_x_bf16, dim3(xn / 8 / 256), dim3(256), 0,
                           stream, x, xbf, xn);
        hipLaunchKernelGGL((dequant_gemm_wmma<true>), grid, block, 0, stream,
                           x, xbf, labels, centroids, bias, out);
    } else {
        hipLaunchKernelGGL((dequant_gemm_wmma<false>), grid, block, 0, stream,
                           x, (const unsigned short*)nullptr, labels,
                           centroids, bias, out);
    }
}
